// CausalSelfAttention_65309272703394
// MI455X (gfx1250) — matrix operand order repaired and checked by reading, not compiled, not run on hardware
//
#include <hip/hip_runtime.h>
#include <math.h>

typedef _Float16 v16h __attribute__((ext_vector_type(16)));
typedef float    v8f  __attribute__((ext_vector_type(8)));

union AF { v16h v; unsigned u[8]; _Float16 h[16]; };

#define WMMA_F16(a, b, c) \
  __builtin_amdgcn_wmma_f32_16x16x32_f16(false, (a), false, (b), (short)0, (c), false, false)

// ---------------------------------------------------------------------------
// Fragment loaders (ISA 7.12.2 layouts, wave32)
// A 16x32 f16: lane l -> m = l&15, half = l>>4; VGPR j holds K pair:
//   j<4 : K = half*8 + 2j ;  j>=4 : K = 16 + half*8 + 2(j-4)
// ---------------------------------------------------------------------------
__device__ __forceinline__ v16h load_afrag(const _Float16* base, int ld) {
  int lane = threadIdx.x & 31;
  int hlf = lane >> 4, m = lane & 15;
  AF f;
  const _Float16* row = base + (size_t)m * ld;
#pragma unroll
  for (int j = 0; j < 8; ++j) {
    int k = (j < 4) ? (hlf * 8 + 2 * j) : (16 + hlf * 8 + 2 * (j - 4));
    f.u[j] = *reinterpret_cast<const unsigned*>(row + k);
  }
  return f.v;
}

// B 32x16 f16 with B[k][n] = W[n0+n][k0+k] (W row-major, base = &W[n0][k0]):
// lane l -> n = l&15, kbase = (l>>4)*8 ; VGPR j holds K = kbase+2j (j<4), 16+kbase+2(j-4) (j>=4), +1
__device__ __forceinline__ v16h load_bfrag(const _Float16* base, int ld) {
  int lane = threadIdx.x & 31;
  int n = lane & 15, kb = (lane >> 4) * 8;
  AF f;
  const _Float16* row = base + (size_t)n * ld + kb;
#pragma unroll
  for (int j = 0; j < 8; ++j)
    f.u[j] = *reinterpret_cast<const unsigned*>(row + ((j < 4) ? 2 * j : 16 + 2 * (j - 4)));
  return f.v;
}

// ---------------------------------------------------------------------------
// Async staging of a 32x128 f16 tile (8KB) into LDS via CDNA5 async engine.
// Each issue: 32 lanes x 16B = 512B ; 16 issues per tile. Tracked by ASYNCcnt.
// ---------------------------------------------------------------------------
__device__ __forceinline__ void stage_tile_async(const _Float16* gsrc, int ldg,
                                                 unsigned ldsbase) {
  int lane = threadIdx.x & 31;
  int cb = (lane & 15) * 8;   // f16 column offset (16B granules)
  int rh = lane >> 4;
#pragma unroll
  for (int t = 0; t < 16; ++t) {
    int r = 2 * t + rh;
    unsigned loff = ldsbase + (unsigned)((r * 128 + cb) * 2);
    unsigned long long ga =
        (unsigned long long)(uintptr_t)(gsrc + (size_t)r * ldg + cb);
    asm volatile("global_load_async_to_lds_b128 %0, %1, off"
                 :: "v"(loff), "v"(ga) : "memory");
  }
}

// ---------------------------------------------------------------------------
// C = A(MxK f16) * W^T (W: NxK f16), C f32. 32x64 tile per wave:
// 2 A-frags + 4 B-frags -> 8 WMMAs per 32-wide k-step (~22 flop/byte).
// ---------------------------------------------------------------------------
__global__ void wmma_gemm_nt(const _Float16* __restrict__ A,
                             const _Float16* __restrict__ W,
                             float* __restrict__ C, int M, int N, int K) {
  (void)M;
  int col0 = blockIdx.x * 64;
  int row0 = blockIdx.y * 32;
  v8f zero = {};
  v8f acc[2][4];
#pragma unroll
  for (int mi = 0; mi < 2; ++mi)
#pragma unroll
    for (int ni = 0; ni < 4; ++ni) acc[mi][ni] = zero;

  const _Float16* a0p = A + (size_t)row0 * K;
  const _Float16* a1p = A + (size_t)(row0 + 16) * K;
  for (int kk = 0; kk < K; kk += 32) {
    __builtin_prefetch(a0p + kk + 512, 0, 1);
    __builtin_prefetch(W + (size_t)col0 * K + kk + 512, 0, 1);
    v16h a0 = load_afrag(a0p + kk, K);
    v16h a1 = load_afrag(a1p + kk, K);
#pragma unroll
    for (int ni = 0; ni < 4; ++ni) {
      v16h b = load_bfrag(W + (size_t)(col0 + ni * 16) * K + kk, K);
      acc[0][ni] = WMMA_F16(a0, b, acc[0][ni]);
      acc[1][ni] = WMMA_F16(a1, b, acc[1][ni]);
    }
  }
  int lane = threadIdx.x & 31;
  int n = lane & 15, g = lane >> 4;
#pragma unroll
  for (int mi = 0; mi < 2; ++mi)
#pragma unroll
    for (int ni = 0; ni < 4; ++ni)
#pragma unroll
      for (int i = 0; i < 8; ++i)
        C[(size_t)(row0 + mi * 16 + i + 8 * g) * N + col0 + ni * 16 + n] =
            acc[mi][ni][i];
}

// ---------------------------------------------------------------------------
__global__ void f32_to_f16(const float* __restrict__ src,
                           _Float16* __restrict__ dst, int n) {
  int i = blockIdx.x * blockDim.x + threadIdx.x;
  int stride = gridDim.x * blockDim.x;
  for (; i < n; i += stride) dst[i] = (_Float16)src[i];
}

// ---------------------------------------------------------------------------
// RMS-norm (+eps) -> RoPE -> gain, per (s, head) row of 128. One wave per row.
// Lane holds d = {lane, lane+32, lane+64, lane+96}; RoPE pairs d <-> d+64.
// ---------------------------------------------------------------------------
__global__ void rmsnorm_rope(const float* __restrict__ src,
                             _Float16* __restrict__ dst,
                             const float* __restrict__ gain, int nh, int S) {
  int wid = (blockIdx.x * blockDim.x + threadIdx.x) >> 5;
  int lane = threadIdx.x & 31;
  if (wid >= S * nh) return;
  int s = wid / nh, h = wid % nh;
  const float* row = src + ((size_t)s * nh + h) * 128;
  float v0 = row[lane], v1 = row[lane + 32], v2 = row[lane + 64], v3 = row[lane + 96];
  float ss = v0 * v0 + v1 * v1 + v2 * v2 + v3 * v3;
#pragma unroll
  for (int off = 1; off < 32; off <<= 1) ss += __shfl_xor(ss, off, 32);
  float r = rsqrtf(ss * (1.0f / 128.0f) + 1.1920929e-07f);
  v0 *= r; v1 *= r; v2 *= r; v3 *= r;
  const float kLn = 0.14391156f;  // ln(10000)/64
  float a0 = (float)s * __expf(-(float)lane * kLn);
  float a1 = (float)s * __expf(-(float)(lane + 32) * kLn);
  float c0, s0, c1, s1;
  sincosf(a0, &s0, &c0);
  sincosf(a1, &s1, &c1);
  float o0 = v0 * c0 + v2 * s0;
  float o2 = -v0 * s0 + v2 * c0;
  float o1 = v1 * c1 + v3 * s1;
  float o3 = -v1 * s1 + v3 * c1;
  float g = gain ? gain[h] : 1.0f;
  _Float16* drow = dst + ((size_t)s * nh + h) * 128;
  drow[lane]      = (_Float16)(o0 * g);
  drow[lane + 32] = (_Float16)(o1 * g);
  drow[lane + 64] = (_Float16)(o2 * g);
  drow[lane + 96] = (_Float16)(o3 * g);
}

// ---------------------------------------------------------------------------
// v outputs: vout (KV,S,hd) f32 to d_out, and transposed f16 Vt (KV,hd,S).
// ---------------------------------------------------------------------------
__global__ void v_post(const float* __restrict__ vbuf, float* __restrict__ vout,
                       _Float16* __restrict__ vt, int S) {
  int i = blockIdx.x * blockDim.x + threadIdx.x;
  if (i >= S * 512) return;
  int s = i >> 9, col = i & 511, kv = col >> 7, d = col & 127;
  float val = vbuf[i];
  vout[((size_t)kv * S + s) * 128 + d] = val;
  vt[((size_t)kv * 128 + d) * S + s] = (_Float16)val;
}

// ---------------------------------------------------------------------------
// Flash attention + v-orthogonalization epilogue.
// One wave per (16-row query tile, head); 2 waves/block.
// K tiles are double-buffered into LDS with global_load_async_to_lds_b128,
// fenced by s_wait_asynccnt, so the next key block streams in during WMMAs.
// ---------------------------------------------------------------------------
__global__ void flash_attn(const _Float16* __restrict__ qh,
                           const _Float16* __restrict__ kh,
                           const _Float16* __restrict__ vt,
                           const float* __restrict__ vbuf,
                           _Float16* __restrict__ yh, int S) {
  __shared__ __align__(16) _Float16 ktile[2][2][32][128];  // [wave][buf] 8KB tiles
  __shared__ __align__(16) _Float16 pT[2][16][32];
  const int H = 16, KV = 4;
  const int D = H * 128, KD = KV * 128;
  const float scale = 0.08838834764831845f;  // 1/sqrt(128)

  int wave = threadIdx.x >> 5, lane = threadIdx.x & 31;
  int qt = blockIdx.x * 2 + wave;
  int h = blockIdx.y;
  int m0 = qt * 16, kvh = h >> 2;
  int g = lane >> 4, n = lane & 15;

  v16h qa[4];
#pragma unroll
  for (int t = 0; t < 4; ++t)
    qa[t] = load_afrag(qh + (size_t)m0 * D + h * 128 + t * 32, D);

  float mrow[8], lrow[8];
  v8f zero = {};
  v8f acc[8];
#pragma unroll
  for (int c = 0; c < 8; ++c) acc[c] = zero;
#pragma unroll
  for (int i = 0; i < 8; ++i) { mrow[i] = -__builtin_inff(); lrow[i] = 0.0f; }

  const _Float16* vtb = vt + (size_t)kvh * 128 * S;
  const _Float16* ksrc = kh + kvh * 128;
  unsigned kbase0 = (unsigned)(uintptr_t)&ktile[wave][0][0][0];
  unsigned kbase1 = (unsigned)(uintptr_t)&ktile[wave][1][0][0];

  int nblocks = m0 / 32 + 1;
  stage_tile_async(ksrc, KD, kbase0);
  asm volatile("s_wait_asynccnt 0x0" ::: "memory");

  int buf = 0;
  for (int kb = 0; kb < nblocks; ++kb) {
    int key0 = kb * 32;
    if (kb + 1 < nblocks)  // prefetch next K tile into the other buffer
      stage_tile_async(ksrc + (size_t)(key0 + 32) * KD, KD, buf ? kbase0 : kbase1);

    const _Float16* kt = &ktile[wave][buf][0][0];
    // ---- scores: 16x32 block of q.k^T (K B-frags from LDS tile) ----
    v8f s0 = zero, s1 = zero;
#pragma unroll
    for (int t = 0; t < 4; ++t) {
      v16h b0 = load_bfrag(kt + t * 32, 128);
      s0 = WMMA_F16(qa[t], b0, s0);
    }
#pragma unroll
    for (int t = 0; t < 4; ++t) {
      v16h b1 = load_bfrag(kt + 16 * 128 + t * 32, 128);
      s1 = WMMA_F16(qa[t], b1, s1);
    }
    bool edge = (key0 + 31 > m0);
#pragma unroll
    for (int i = 0; i < 8; ++i) {
      int r = m0 + i + 8 * g;
      float a = s0[i] * scale, b = s1[i] * scale;
      if (edge) {
        if (key0 + n > r)      a = -__builtin_inff();
        if (key0 + 16 + n > r) b = -__builtin_inff();
      }
      s0[i] = a; s1[i] = b;
    }
    // ---- online softmax update ----
#pragma unroll
    for (int i = 0; i < 8; ++i) {
      float loc = fmaxf(s0[i], s1[i]);
#pragma unroll
      for (int off = 1; off < 16; off <<= 1) loc = fmaxf(loc, __shfl_xor(loc, off, 32));
      float mn = fmaxf(mrow[i], loc);
      float alpha = __expf(mrow[i] - mn);
      float p0 = __expf(s0[i] - mn);
      float p1 = __expf(s1[i] - mn);
      float rs = p0 + p1;
#pragma unroll
      for (int off = 1; off < 16; off <<= 1) rs += __shfl_xor(rs, off, 32);
      lrow[i] = lrow[i] * alpha + rs;
      mrow[i] = mn;
#pragma unroll
      for (int c = 0; c < 8; ++c) acc[c][i] *= alpha;
      pT[wave][i + 8 * g][n]      = (_Float16)p0;
      pT[wave][i + 8 * g][n + 16] = (_Float16)p1;
    }
    // same-wave LDS store->load ordering (CDNA5 split DS counter)
    asm volatile("s_wait_dscnt 0x0" ::: "memory");
    // ---- re-read P in A-fragment layout ----
    AF pf;
    {
      int hlf = lane >> 4, mm = lane & 15;
#pragma unroll
      for (int j = 0; j < 8; ++j) {
        int k = (j < 4) ? (hlf * 8 + 2 * j) : (16 + hlf * 8 + 2 * (j - 4));
        pf.u[j] = *reinterpret_cast<const unsigned*>(&pT[wave][mm][k]);
      }
    }
    // ---- acc += P (16x32) @ V (32x128), 8 N-chunks (Vt contiguous) ----
#pragma unroll
    for (int c = 0; c < 8; ++c) {
      v16h vb = load_bfrag(vtb + (size_t)(c * 16) * S + key0, S);
      acc[c] = WMMA_F16(pf.v, vb, acc[c]);
    }
    // ensure the prefetched K tile has landed before switching buffers
    asm volatile("s_wait_asynccnt 0x0" ::: "memory");
    buf ^= 1;
  }

  // ---- epilogue: y = acc/l ; y -= (y.vn) vn ; store f16 ----
#pragma unroll
  for (int i = 0; i < 8; ++i) {
    int r = m0 + i + 8 * g;
    float inv = 1.0f / lrow[i];
    float yv[8], vr[8];
    float dyv = 0.0f, dvv = 0.0f;
    const float* vrow = vbuf + (size_t)r * KD + kvh * 128;
#pragma unroll
    for (int c = 0; c < 8; ++c) {
      yv[c] = acc[c][i] * inv;
      vr[c] = vrow[c * 16 + n];
      dyv += yv[c] * vr[c];
      dvv += vr[c] * vr[c];
    }
#pragma unroll
    for (int off = 1; off < 16; off <<= 1) {
      dyv += __shfl_xor(dyv, off, 32);
      dvv += __shfl_xor(dvv, off, 32);
    }
    float nv = fmaxf(sqrtf(dvv), 1e-12f);
    float coef = dyv / (nv * nv);
    _Float16* drow = yh + (size_t)r * D + h * 128;
#pragma unroll
    for (int c = 0; c < 8; ++c)
      drow[c * 16 + n] = (_Float16)(yv[c] - coef * vr[c]);
  }
}

// ---------------------------------------------------------------------------
extern "C" void kernel_launch(void* const* d_in, const int* in_sizes, int n_in,
                              void* d_out, int out_size, void* d_ws, size_t ws_size,
                              hipStream_t stream) {
  (void)in_sizes; (void)n_in; (void)out_size; (void)ws_size;
  const int S = 2048, D = 2048, H = 16, KV = 4, KD = KV * 128;

  const float* x  = (const float*)d_in[0];
  const float* Wq = (const float*)d_in[1];
  const float* Wk = (const float*)d_in[2];
  const float* Wv = (const float*)d_in[3];
  const float* Wp = (const float*)d_in[4];
  const float* qg = (const float*)d_in[5];

  float* outp = (float*)d_out;            // (S, D)
  float* vout = outp + (size_t)S * D;     // (KV, S, 128)

  char* ws = (char*)d_ws;
  size_t off = 0;
  auto carve = [&](size_t bytes) {
    void* p = ws + off;
    off += (bytes + 255) & ~(size_t)255;
    return p;
  };
  _Float16* xh  = (_Float16*)carve((size_t)S * D * 2);
  _Float16* Wqh = (_Float16*)carve((size_t)D * D * 2);
  _Float16* Wkh = (_Float16*)carve((size_t)KD * D * 2);
  _Float16* Wvh = (_Float16*)carve((size_t)KD * D * 2);
  _Float16* Wph = (_Float16*)carve((size_t)D * D * 2);
  float*    qb  = (float*)carve((size_t)S * D * 4);
  float*    kb  = (float*)carve((size_t)S * KD * 4);
  float*    vb  = (float*)carve((size_t)S * KD * 4);
  _Float16* qhb = (_Float16*)carve((size_t)S * D * 2);
  _Float16* khb = (_Float16*)carve((size_t)S * KD * 2);
  _Float16* vtb = (_Float16*)carve((size_t)S * KD * 2);
  _Float16* yhb = (_Float16*)carve((size_t)S * D * 2);

  // 1. f32 -> f16 conversions
  f32_to_f16<<<4096, 256, 0, stream>>>(x,  xh,  S * D);
  f32_to_f16<<<4096, 256, 0, stream>>>(Wq, Wqh, D * D);
  f32_to_f16<<<1024, 256, 0, stream>>>(Wk, Wkh, KD * D);
  f32_to_f16<<<1024, 256, 0, stream>>>(Wv, Wvh, KD * D);
  f32_to_f16<<<4096, 256, 0, stream>>>(Wp, Wph, D * D);

  // 2. QKV projections (WMMA, 32x64 tile per wave)
  wmma_gemm_nt<<<dim3(D / 64,  S / 32), 32, 0, stream>>>(xh, Wqh, qb, S, D,  D);
  wmma_gemm_nt<<<dim3(KD / 64, S / 32), 32, 0, stream>>>(xh, Wkh, kb, S, KD, D);
  wmma_gemm_nt<<<dim3(KD / 64, S / 32), 32, 0, stream>>>(xh, Wvh, vb, S, KD, D);

  // 3. RMS-norm + RoPE (+gain on q)
  {
    int qwaves = S * H;
    rmsnorm_rope<<<(qwaves + 3) / 4, 128, 0, stream>>>(qb, qhb, qg, H, S);
    int kwaves = S * KV;
    rmsnorm_rope<<<(kwaves + 3) / 4, 128, 0, stream>>>(kb, khb, nullptr, KV, S);
  }

  // 4. v output + transposed f16 copy for P@V fragments
  v_post<<<(S * KD + 255) / 256, 256, 0, stream>>>(vb, vout, vtb, S);

  // 5. causal flash attention, async-LDS double-buffered K tiles
  flash_attn<<<dim3(S / 32, H), 64, 0, stream>>>(qhb, khb, vtb, vb, yhb, S);

  // 6. output projection (WMMA)
  wmma_gemm_nt<<<dim3(D / 64, S / 32), 32, 0, stream>>>(yhb, Wph, outp, S, D, D);
}
